// CrossViewRegionAligner_16939351015836
// MI455X (gfx1250) — matrix operand order repaired and checked by reading, not compiled, not run on hardware
//
#include <hip/hip_runtime.h>
#include <hip/hip_bf16.h>

typedef __attribute__((ext_vector_type(16))) _Float16 v16h;
typedef __attribute__((ext_vector_type(8)))  _Float16 h8;
typedef __attribute__((ext_vector_type(8)))  float    v8f;

typedef unsigned int u32x4 __attribute__((ext_vector_type(4)));
typedef int          i32x8 __attribute__((ext_vector_type(8)));
typedef int          i32x4 __attribute__((ext_vector_type(4)));

#define N_SAT 1536
#define N_UAV 1536
#define DDIM  64

#if defined(__gfx1250__) && __has_builtin(__builtin_amdgcn_tensor_load_to_lds)
#define USE_TDM 1
#else
#define USE_TDM 0
#endif

// ---------------------------------------------------------------------------
// Kernel A: hs = sat @ W1[:64] + b1 (f16), hu = uav @ W1[64:] (f16)
// One wave per 16-row M-tile; K=64 in two x32 steps; N=64 in four 16-col tiles.
// ---------------------------------------------------------------------------
__global__ __launch_bounds__(32)
void k_embed(const float* __restrict__ sat, const float* __restrict__ uav,
             const float* __restrict__ W1, const float* __restrict__ b1,
             _Float16* __restrict__ hs, _Float16* __restrict__ hu)
{
    const int lane = threadIdx.x & 31;
    const int m    = lane & 15;
    const int hi   = lane >> 4;
    const int mat  = blockIdx.y;          // 0 = sat, 1 = uav
    const int i0   = blockIdx.x * 16;
    const float*  X = mat ? uav : sat;
    _Float16*     O = mat ? hu  : hs;

    // B fragments from W1 (K = input dim, N = output dim), column-strided reads
    v16h B[2][4];
    for (int kk = 0; kk < 2; ++kk)
        for (int nt = 0; nt < 4; ++nt) {
            const int kbase = 64*mat + 32*kk + 16*hi;
            const int n     = 16*nt + m;
            v16h f;
            #pragma unroll
            for (int e = 0; e < 16; ++e)
                f[e] = (_Float16)W1[(size_t)(kbase + e)*DDIM + n];
            B[kk][nt] = f;
        }

    // A fragments: lane holds two runs of 8 consecutive K per 16x32 step
    v16h A[2];
    const float* row = X + (size_t)(i0 + m)*DDIM;
    for (int kk = 0; kk < 2; ++kk) {
        const int s0 = 32*kk + 16*hi;
        const int s1 = s0 + 8;
        v16h f;
        #pragma unroll
        for (int e = 0; e < 8; ++e) {
            f[e]     = (_Float16)row[s0 + e];
            f[8 + e] = (_Float16)row[s1 + e];
        }
        A[kk] = f;
    }

    v8f acc[4];
    for (int nt = 0; nt < 4; ++nt) {
        #pragma unroll
        for (int r = 0; r < 8; ++r) acc[nt][r] = 0.f;
    }
    for (int nt = 0; nt < 4; ++nt) {
        v8f c = acc[nt];
        c = __builtin_amdgcn_wmma_f32_16x16x32_f16(false, A[0], false, B[0][nt],
                                                   (short)0, c, false, false);
        c = __builtin_amdgcn_wmma_f32_16x16x32_f16(false, A[1], false, B[1][nt],
                                                   (short)0, c, false, false);
        acc[nt] = c;
    }

    // C/D layout: lane L VGPR r -> (M = r + 8*hi, N = L%16). Fold b1 into hs.
    for (int nt = 0; nt < 4; ++nt) {
        const int n = 16*nt + m;
        const float bias = (mat == 0) ? b1[n] : 0.f;
        #pragma unroll
        for (int r = 0; r < 8; ++r)
            O[(size_t)(i0 + r + 8*hi)*DDIM + n] = (_Float16)(acc[nt][r] + bias);
    }
}

// ---------------------------------------------------------------------------
// Kernel B (fused pair MLP): never materializes h1[N,M,64].
// Block = 4 waves; block owns one 16-uav tile (TDM-staged to LDS), each wave
// owns a 16-sat tile. Per uav j: A = relu(hs_tile + hu_j) built in registers,
// then 4x v_wmma_f32_16x16x32_f16 -> h2[16x32], relu, dot W3 (shfl_xor),
// sigmoid -> sim.
// ---------------------------------------------------------------------------
__global__ __launch_bounds__(128)
void k_score(const _Float16* __restrict__ hs, const _Float16* __restrict__ hu,
             const float* __restrict__ W2, const float* __restrict__ b2,
             const float* __restrict__ W3, const float* __restrict__ b3,
             float* __restrict__ sim)
{
    __shared__ _Float16 sHU[16 * DDIM];

    const int t    = threadIdx.x;
    const int lane = t & 31;
    const int wave = t >> 5;
    const int m    = lane & 15;
    const int hi   = lane >> 4;
    const int j0   = blockIdx.x * 16;
    const int i0   = blockIdx.y * 64 + wave * 16;

    __builtin_prefetch(W2, 0, 3);   // global_prefetch_b8 of the shared weights

#if USE_TDM
    // Tensor Data Mover: DMA the 16x64 f16 hu tile into LDS (TENSORcnt path).
    if (wave == 0) {
        const unsigned long long ga =
            (unsigned long long)(uintptr_t)(hu + (size_t)j0 * DDIM);
        const unsigned int ldsOff = (unsigned int)(uintptr_t)&sHU[0];
        // D# group0: count=1 | lds_addr | global_addr[56:0] | type=2
        u32x4 g0 = { 1u, ldsOff, (unsigned int)ga,
                     (unsigned int)((ga >> 32) & 0x01ffffffu) | 0x80000000u };
        // D# group1: data_size=2B; tensor_dim0=64; tensor_dim1=1536;
        //            tile_dim0=64; tile_dim1=16; tensor_dim0_stride=64
        i32x8 g1 = { (int)(1u << 16),              // workgroup_mask=0, data_size=1(2B)
                     (int)(64u << 16),             // abar_addr=0 | tensor_dim0 lo
                     (int)((unsigned)N_UAV << 16), // tensor_dim0 hi=0 | tensor_dim1 lo
                     (int)(64u << 16),             // tensor_dim1 hi=0 | tile_dim0=64
                     16,                           // tile_dim1=16, tile_dim2=0
                     64,                           // tensor_dim0_stride lo = 64
                     0, 0 };                       // stride0 hi, tensor_dim1_stride
        i32x4 g2 = { 0, 0, 0, 0 };                 // 2D tensor: groups 2/3 unused
        i32x4 g3 = { 0, 0, 0, 0 };
        i32x8 g4 = { 0, 0, 0, 0, 0, 0, 0, 0 };     // extra group (clang-23 6-arg form)
        __builtin_amdgcn_tensor_load_to_lds(g0, g1, g2, g3, g4, 0);
        __builtin_amdgcn_s_wait_tensorcnt(0);
    }
    __syncthreads();
#else
    // Fallback: cooperative staging of the 16x64 f16 hu tile (128 x 16B)
    {
        const int row = t >> 3, seg = t & 7;
        *(h8*)&sHU[row*DDIM + seg*8] = *(const h8*)&hu[(size_t)(j0+row)*DDIM + seg*8];
    }
    __syncthreads();
#endif

    // B fragments from W2 [64 x 32]
    v16h B[2][2];
    for (int kk = 0; kk < 2; ++kk)
        for (int nt = 0; nt < 2; ++nt) {
            const int kbase = 32*kk + 16*hi;
            const int n     = 16*nt + m;
            v16h f;
            #pragma unroll
            for (int e = 0; e < 16; ++e)
                f[e] = (_Float16)W2[(size_t)(kbase + e)*32 + n];
            B[kk][nt] = f;
        }

    // hs fragments for this wave's 16 sat rows (b1 already folded in)
    v16h S[2];
    {
        const _Float16* row = hs + (size_t)(i0 + m)*DDIM;
        for (int kk = 0; kk < 2; ++kk) {
            const int s0 = 32*kk + 16*hi, s1 = s0 + 8;
            union { v16h v; h8 h[2]; } u;
            u.h[0] = *(const h8*)(row + s0);
            u.h[1] = *(const h8*)(row + s1);
            S[kk] = u.v;
        }
    }

    const float w3lo = W3[m];
    const float w3hi = W3[16 + m];
    const float b3s  = b3[0];
    const float c0   = b2[m];        // accumulator init = b2 (per-lane column)
    const float c1   = b2[16 + m];

    for (int jj = 0; jj < 16; ++jj) {
        // A = relu(hs + hu_j) assembled straight into WMMA layout
        v16h Af[2];
        const _Float16* ur = &sHU[jj * DDIM];
        for (int kk = 0; kk < 2; ++kk) {
            const int s0 = 32*kk + 16*hi, s1 = s0 + 8;
            union { v16h v; h8 h[2]; } u;
            u.h[0] = *(const h8*)(ur + s0);      // ds_load_b128
            u.h[1] = *(const h8*)(ur + s1);
            v16h a = S[kk] + u.v;
            #pragma unroll
            for (int e = 0; e < 16; ++e)
                a[e] = a[e] > (_Float16)0.f ? a[e] : (_Float16)0.f;
            Af[kk] = a;
        }

        v8f acc0, acc1;
        #pragma unroll
        for (int r = 0; r < 8; ++r) { acc0[r] = c0; acc1[r] = c1; }
        acc0 = __builtin_amdgcn_wmma_f32_16x16x32_f16(false, Af[0], false, B[0][0],
                                                      (short)0, acc0, false, false);
        acc0 = __builtin_amdgcn_wmma_f32_16x16x32_f16(false, Af[1], false, B[1][0],
                                                      (short)0, acc0, false, false);
        acc1 = __builtin_amdgcn_wmma_f32_16x16x32_f16(false, Af[0], false, B[0][1],
                                                      (short)0, acc1, false, false);
        acc1 = __builtin_amdgcn_wmma_f32_16x16x32_f16(false, Af[1], false, B[1][1],
                                                      (short)0, acc1, false, false);

        // relu(h2) . W3  -> per-lane partial over N, reduce across 16-lane group
        float p[8];
        #pragma unroll
        for (int r = 0; r < 8; ++r) {
            float ha = acc0[r] > 0.f ? acc0[r] : 0.f;
            float hb = acc1[r] > 0.f ? acc1[r] : 0.f;
            p[r] = ha * w3lo + hb * w3hi;
        }
        #pragma unroll
        for (int off = 1; off < 16; off <<= 1) {
            #pragma unroll
            for (int r = 0; r < 8; ++r)
                p[r] += __shfl_xor(p[r], off, 16);
        }
        if (m == 0) {
            #pragma unroll
            for (int r = 0; r < 8; ++r) {
                float x = p[r] + b3s;
                sim[(size_t)(i0 + r + 8*hi)*N_UAV + (j0 + jj)] =
                    1.f / (1.f + __expf(-x));
            }
        }
    }
}

// ---------------------------------------------------------------------------
// Kernel C: sequential greedy argmax assignment (single workgroup; sim is L2-
// resident at 9.4 MB). Matches jnp.where(mask, row, -1) + first-max argmax.
// ---------------------------------------------------------------------------
__global__ __launch_bounds__(1024)
void k_assign(const float* __restrict__ sim, int* __restrict__ assign)
{
    __shared__ float sBest[1024];
    __shared__ int   sIdx[1024];
    __shared__ int   sMask[N_UAV];
    const int t = threadIdx.x;
    for (int j = t; j < N_UAV; j += 1024) sMask[j] = 1;
    __syncthreads();

    for (int i = 0; i < N_SAT; ++i) {
        const float* row = sim + (size_t)i * N_UAV;
        float best = -2.f; int bidx = 0x7fffffff;
        for (int j = t; j < N_UAV; j += 1024) {
            float v = sMask[j] ? row[j] : -1.f;
            if (v > best || (v == best && j < bidx)) { best = v; bidx = j; }
        }
        sBest[t] = best; sIdx[t] = bidx;
        __syncthreads();
        for (int s = 512; s > 0; s >>= 1) {
            if (t < s) {
                float vb = sBest[t + s]; int vi = sIdx[t + s];
                if (vb > sBest[t] || (vb == sBest[t] && vi < sIdx[t])) {
                    sBest[t] = vb; sIdx[t] = vi;
                }
            }
            __syncthreads();
        }
        if (t == 0) { assign[i] = sIdx[0]; sMask[sIdx[0]] = 0; }
        __syncthreads();
    }
}

// ---------------------------------------------------------------------------
// Kernel D: out[i] = (sat[i], uav[assign[i]])  -> [N, 2, 64] f32
// ---------------------------------------------------------------------------
__global__ __launch_bounds__(128)
void k_gather(const float* __restrict__ sat, const float* __restrict__ uav,
              const int* __restrict__ assign, float* __restrict__ out)
{
    const int i = blockIdx.x;
    const int t = threadIdx.x;
    if (t < 64)
        out[(size_t)i*128 + t] = sat[(size_t)i*DDIM + t];
    else
        out[(size_t)i*128 + t] = uav[(size_t)assign[i]*DDIM + (t - 64)];
}

// ---------------------------------------------------------------------------
extern "C" void kernel_launch(void* const* d_in, const int* in_sizes, int n_in,
                              void* d_out, int out_size, void* d_ws, size_t ws_size,
                              hipStream_t stream)
{
    const float* sat = (const float*)d_in[0];
    const float* uav = (const float*)d_in[1];
    const float* W1  = (const float*)d_in[2];
    const float* b1  = (const float*)d_in[3];
    const float* W2  = (const float*)d_in[4];
    const float* b2  = (const float*)d_in[5];
    const float* W3  = (const float*)d_in[6];
    const float* b3  = (const float*)d_in[7];
    float* out = (float*)d_out;

    // Workspace layout: hs f16 | hu f16 | sim f32 | assign i32  (~9.84 MB)
    _Float16* hs = (_Float16*)d_ws;
    _Float16* hu = hs + (size_t)N_SAT * DDIM;
    float*   sim = (float*)(hu + (size_t)N_UAV * DDIM);
    int*  assign = (int*)(sim + (size_t)N_SAT * N_UAV);

    k_embed <<<dim3(N_SAT/16, 2),        32,  0, stream>>>(sat, uav, W1, b1, hs, hu);
    k_score <<<dim3(N_UAV/16, N_SAT/64), 128, 0, stream>>>(hs, hu, W2, b2, W3, b3, sim);
    k_assign<<<1,                        1024,0, stream>>>(sim, assign);
    k_gather<<<N_SAT,                    128, 0, stream>>>(sat, uav, assign, out);
}
